// CascadeTransformer_68135361184402
// MI455X (gfx1250) — matrix operand order repaired and checked by reading, not compiled, not run on hardware
//
#include <hip/hip_runtime.h>
#include <hip/hip_bf16.h>
#include <math.h>

// ---------------------------------------------------------------------------
// CDNA5 (gfx1250) transformer forward: bf16 WMMA + async global->LDS pipeline.
//   V=32000 D=1024 H=16 DK=64 DFF=4096 L=6 B=2 S=2048  (M = B*S = 4096 tokens)
// ---------------------------------------------------------------------------

typedef __bf16 bf16;
typedef __bf16 bf16x8  __attribute__((ext_vector_type(8)));
typedef __bf16 bf16x16 __attribute__((ext_vector_type(16)));
typedef float  f32x8   __attribute__((ext_vector_type(8)));

#define DEV __device__ __forceinline__

DEV bf16x16 cat8(bf16x8 lo, bf16x8 hi) {
    return __builtin_shufflevector(lo, hi, 0,1,2,3,4,5,6,7,8,9,10,11,12,13,14,15);
}

// D(16x16 f32) = A(16x32 bf16) * B(32x16 bf16) + C
DEV f32x8 bwmma(bf16x16 a, bf16x16 b, f32x8 c) {
    return __builtin_amdgcn_wmma_f32_16x16x32_bf16(
        /*neg_a=*/false, a, /*neg_b=*/false, b,
        /*c_mod=*/(short)0, c, /*reuse_a=*/false, /*reuse_b=*/false);
}

// 16-byte async copy global -> LDS (ASYNCcnt-tracked, ISA 08_async_tensor §4).
// ldsOff: LDS byte offset (low 32 bits of a flat LDS pointer); g: global src.
DEV void async_cp16(uint32_t ldsOff, const bf16* g) {
    asm volatile("global_load_async_to_lds_b128 %0, %1, off"
                 :: "v"(ldsOff), "v"(g) : "memory");
}
DEV void async_wait0() {
    asm volatile("s_wait_asynccnt 0x0" ::: "memory");
}

// ---------------------------------------------------------------------------
// Weight convert + transpose:  in f32 [nmat][K][N]  ->  out bf16 [nmat][N][K]
// (so the GEMM B-operand reads are contiguous per lane)
// ---------------------------------------------------------------------------
__global__ void cvt_transpose_kernel(const float* __restrict__ in,
                                     bf16* __restrict__ out,
                                     int K, int N, long total) {
    long tid = (long)blockIdx.x * blockDim.x + threadIdx.x;
    if (tid >= total) return;
    long per = (long)K * N;
    long m   = tid / per;
    long rem = tid - m * per;
    long n   = rem / K;
    long k   = rem - n * K;
    out[m * per + n * K + k] = (bf16)in[m * per + k * N + n];
}

// ---------------------------------------------------------------------------
// Embedding gather + sinusoidal positional encoding (computed on the fly)
// ---------------------------------------------------------------------------
__global__ __launch_bounds__(256)
void embed_pe_kernel(const int* __restrict__ ids, const float* __restrict__ emb,
                     float* __restrict__ x, int Sdim, int Ddim) {
    int token = blockIdx.x;            // b*S + s
    int s     = token % Sdim;
    const float* e = emb + (size_t)ids[token] * Ddim;
    float*      xr = x   + (size_t)token * Ddim;
    const float neg_log1e4 = -9.210340371976184f;  // -ln(10000)
    for (int d = threadIdx.x; d < Ddim; d += blockDim.x) {
        int   i2  = d & ~1;
        float div = __expf(neg_log1e4 * (float)i2 / (float)Ddim);
        float ang = (float)s * div;
        float pe  = (d & 1) ? __cosf(ang) : __sinf(ang);
        xr[d] = e[d] + pe;
    }
}

// ---------------------------------------------------------------------------
// LayerNorm over D=1024, f32 in -> bf16 out (next GEMM's A operand)
// ---------------------------------------------------------------------------
__global__ __launch_bounds__(256)
void layernorm_bf16_kernel(const float* __restrict__ x,
                           const float* __restrict__ w,
                           const float* __restrict__ bia,
                           bf16* __restrict__ out, int Ddim) {
    __shared__ float sSum[256];
    __shared__ float sSq[256];
    int token = blockIdx.x;
    const float* xr = x + (size_t)token * Ddim;
    float s = 0.f, q = 0.f;
    for (int d = threadIdx.x; d < Ddim; d += 256) { float v = xr[d]; s += v; q += v * v; }
    sSum[threadIdx.x] = s; sSq[threadIdx.x] = q;
    __syncthreads();
    for (int off = 128; off > 0; off >>= 1) {
        if (threadIdx.x < off) {
            sSum[threadIdx.x] += sSum[threadIdx.x + off];
            sSq[threadIdx.x]  += sSq[threadIdx.x + off];
        }
        __syncthreads();
    }
    float mu  = sSum[0] / (float)Ddim;
    float var = sSq[0] / (float)Ddim - mu * mu;
    float rs  = rsqrtf(var + 1e-5f);
    bf16* o = out + (size_t)token * Ddim;
    for (int d = threadIdx.x; d < Ddim; d += 256)
        o[d] = (bf16)(((xr[d] - mu) * rs) * w[d] + bia[d]);
}

// ---------------------------------------------------------------------------
// Tiled WMMA GEMM with double-buffered async global->LDS staging.
//   C[MxN] = A[MxK](bf16, row-major) * Bt[NxK](bf16)  + epilogue
// Block = 256 threads = 8 waves; wave tile 32x64; block tile 64x256; K-step 32.
// Per K-step: A tile 64x32 (4KB) + B tile 256x32 (16KB) prefetched with
// GLOBAL_LOAD_ASYNC_TO_LDS_B128 while the previous tile feeds the WMMAs.
// LDS rows padded to 40 elems (80B) -> 16 distinct banks per fragment read.
// All dims used here divide the tiles exactly (no bounds checks).
// ---------------------------------------------------------------------------
enum { EPI_F32 = 0, EPI_BIAS_RES_F32 = 1, EPI_BF16 = 2, EPI_BIAS_GELU_BF16 = 3, EPI_BF16_VT = 4 };

template <int EPI>
__global__ __launch_bounds__(256)
void gemm_bf16_kernel(const bf16* __restrict__ A, const bf16* __restrict__ Bt,
                      int Mdim, int Ndim, int Kdim,
                      float* __restrict__ outF, bf16* __restrict__ outB,
                      const float* __restrict__ bias, const float* __restrict__ res,
                      int S_, int H_, int DK_) {
    constexpr int LP = 40;  // padded LDS row length in elements (80B, 16B-aligned)
    __shared__ __align__(16) bf16 As[2][64][LP];
    __shared__ __align__(16) bf16 Bs[2][256][LP];

    int tid  = threadIdx.x;
    int lane = tid & 31, w = tid >> 5;
    int wm = w >> 2;                 // 0..1 (M sub-tile)
    int wn = w & 3;                  // 0..3 (N sub-tile)
    int m0 = blockIdx.x * 64;
    int n0 = blockIdx.y * 256;
    int lane15 = lane & 15, kh = lane >> 4;

    // Async prefetch of one K-step tile into LDS buffer `buf`.
    auto prefetch = [&](int kk, int buf) {
        {   // A tile: 64 rows x 64B = 256 x 16B transfers (one per thread)
            int r = tid >> 2, c = (tid & 3) * 8;
            async_cp16((uint32_t)(uintptr_t)&As[buf][r][c],
                       A + (size_t)(m0 + r) * Kdim + kk + c);
        }
        // B tile: 256 rows x 64B = 1024 x 16B transfers (four per thread)
        const bf16* g = Bt + (size_t)(n0 + tid) * Kdim + kk;
        uint32_t l0 = (uint32_t)(uintptr_t)&Bs[buf][tid][0];
#pragma unroll
        for (int c = 0; c < 4; ++c) async_cp16(l0 + c * 16, g + c * 8);
    };

    f32x8 acc[2][4] = {};
    int nk = Kdim >> 5;

    prefetch(0, 0);
    async_wait0();
    __syncthreads();

    for (int kt = 0; kt < nk; ++kt) {
        int buf = kt & 1;
        if (kt + 1 < nk) prefetch((kt + 1) << 5, buf ^ 1);

        // A fragments (16x32): per-lane K runs [8*kh,+8) and [16+8*kh,+8)
        int ra0 = wm * 32 + lane15, ra1 = ra0 + 16;
        bf16x16 a0 = cat8(*(const bf16x8*)&As[buf][ra0][8 * kh],
                          *(const bf16x8*)&As[buf][ra0][16 + 8 * kh]);
        bf16x16 a1 = cat8(*(const bf16x8*)&As[buf][ra1][8 * kh],
                          *(const bf16x8*)&As[buf][ra1][16 + 8 * kh]);
#pragma unroll
        for (int t = 0; t < 4; ++t) {
            // B fragment (32x16): lane = column n, K runs [8*kh,+8) and [16+8*kh,+8)
            int rb = wn * 64 + t * 16 + lane15;
            bf16x16 bfrag = cat8(*(const bf16x8*)&Bs[buf][rb][8 * kh],
                                 *(const bf16x8*)&Bs[buf][rb][16 + 8 * kh]);
            acc[0][t] = bwmma(a0, bfrag, acc[0][t]);
            acc[1][t] = bwmma(a1, bfrag, acc[1][t]);
        }
        async_wait0();       // next tile's async LDS writes (this wave) done
        __syncthreads();     // ... and everyone else's, before we read them
    }

#pragma unroll
    for (int i = 0; i < 2; ++i)
#pragma unroll
        for (int t = 0; t < 4; ++t)
#pragma unroll
            for (int r = 0; r < 8; ++r) {
                int m = m0 + wm * 32 + i * 16 + r + 8 * kh;  // C-layout row
                int n = n0 + wn * 64 + t * 16 + lane15;      // C-layout column
                float v = acc[i][t][r];
                if (EPI == EPI_F32) {
                    outF[(size_t)m * Ndim + n] = v;
                } else if (EPI == EPI_BIAS_RES_F32) {
                    outF[(size_t)m * Ndim + n] = v + bias[n] + res[(size_t)m * Ndim + n];
                } else if (EPI == EPI_BF16) {
                    outB[(size_t)m * Ndim + n] = (bf16)v;
                } else if (EPI == EPI_BIAS_GELU_BF16) {
                    float z = v + bias[n];
                    z = 0.5f * z * (1.f + erff(z * 0.70710678118654752f));
                    outB[(size_t)m * Ndim + n] = (bf16)z;
                } else {  // EPI_BF16_VT: scatter V^T as [b][h][dk][s]
                    int b  = m / S_, s = m - b * S_;
                    int h  = n / DK_, dk = n - h * DK_;
                    outB[(((size_t)b * H_ + h) * DK_ + dk) * S_ + s] = (bf16)v;
                }
            }
}

// ---------------------------------------------------------------------------
// Wave-level causal flash attention.
//   Q,K: bf16 [token][D] (slice per head);  Vt: bf16 [b][h][dk][s];
//   O  : bf16 [token][D].
// Each wave owns 16 q-rows; KV tiles of 32 keys; online softmax; P transposed
// C-layout -> A-layout through per-wave LDS (DS ops are in-order per wave).
// ---------------------------------------------------------------------------
__global__ __launch_bounds__(256)
void flash_attn_kernel(const bf16* __restrict__ Q, const bf16* __restrict__ Kb,
                       const bf16* __restrict__ Vt, bf16* __restrict__ O,
                       int Sdim, int Hn, int DKn, int Dn, float scale) {
    __shared__ __align__(16) bf16 pT[8][16][32];
    int lane = threadIdx.x & 31, w = threadIdx.x >> 5;
    int lane15 = lane & 15, kh = lane >> 4;
    int bh = blockIdx.y;
    int b  = bh / Hn, h = bh % Hn;
    int q0 = blockIdx.x * 128 + w * 16;

    const bf16* qrow = Q + (size_t)(b * Sdim + q0 + lane15) * Dn + h * DKn;
    bf16x16 qa0 = cat8(*(const bf16x8*)(qrow + 16 * kh),      *(const bf16x8*)(qrow + 8 + 16 * kh));
    bf16x16 qa1 = cat8(*(const bf16x8*)(qrow + 32 + 16 * kh), *(const bf16x8*)(qrow + 40 + 16 * kh));

    f32x8 o0 = {}, o1 = {}, o2 = {}, o3 = {};
    float rowmax[8], rowsum[8];
#pragma unroll
    for (int r = 0; r < 8; ++r) { rowmax[r] = -1e30f; rowsum[r] = 0.f; }

    const bf16* kbase = Kb + (size_t)b * Sdim * Dn + h * DKn;
    const bf16* vbase = Vt + (size_t)(b * Hn + h) * DKn * Sdim;
    int ntiles = (q0 + 16 + 31) / 32;   // keys 0 .. q0+15 (causal)

    for (int t = 0; t < ntiles; ++t) {
        int kv = t * 32;
        // ---- scores: S(16x32) = Q(16x64) * K^T, two 16-key column sub-tiles
        f32x8 s0 = {}, s1 = {};
        {
            const bf16* kr0 = kbase + (size_t)(kv + lane15) * Dn;
            const bf16* kr1 = kbase + (size_t)(kv + 16 + lane15) * Dn;
            bf16x16 b00 = *(const bf16x16*)(kr0 + 16 * kh);
            bf16x16 b01 = *(const bf16x16*)(kr0 + 32 + 16 * kh);
            bf16x16 b10 = *(const bf16x16*)(kr1 + 16 * kh);
            bf16x16 b11 = *(const bf16x16*)(kr1 + 32 + 16 * kh);
            s0 = bwmma(qa0, b00, s0); s0 = bwmma(qa1, b01, s0);
            s1 = bwmma(qa0, b10, s1); s1 = bwmma(qa1, b11, s1);
        }
        // ---- online softmax (per q-row), causal mask, write P to LDS
#pragma unroll
        for (int r = 0; r < 8; ++r) {
            int qi = q0 + r + 8 * kh;
            float v0 = s0[r] * scale; if (kv + lane15 > qi)      v0 = -1e30f;
            float v1 = s1[r] * scale; if (kv + 16 + lane15 > qi) v1 = -1e30f;
            float mx = fmaxf(v0, v1);
            mx = fmaxf(mx, __shfl_xor(mx, 1, 32));
            mx = fmaxf(mx, __shfl_xor(mx, 2, 32));
            mx = fmaxf(mx, __shfl_xor(mx, 4, 32));
            mx = fmaxf(mx, __shfl_xor(mx, 8, 32));
            float mn = fmaxf(rowmax[r], mx);
            float f  = __expf(rowmax[r] - mn);
            rowmax[r] = mn;
            float p0 = __expf(v0 - mn), p1 = __expf(v1 - mn);
            float ps = p0 + p1;
            ps += __shfl_xor(ps, 1, 32);
            ps += __shfl_xor(ps, 2, 32);
            ps += __shfl_xor(ps, 4, 32);
            ps += __shfl_xor(ps, 8, 32);
            rowsum[r] = rowsum[r] * f + ps;
            o0[r] *= f; o1[r] *= f; o2[r] *= f; o3[r] *= f;
            pT[w][r + 8 * kh][lane15]      = (bf16)p0;
            pT[w][r + 8 * kh][16 + lane15] = (bf16)p1;
        }
        // ---- O(16x64) += P(16x32) * V(32x64); P re-read in A-layout from LDS
        bf16x16 pa = cat8(*(const bf16x8*)&pT[w][lane15][16 * kh],
                          *(const bf16x8*)&pT[w][lane15][16 * kh + 8]);
        const bf16* vp = vbase + kv + 16 * kh;
        o0 = bwmma(pa, *(const bf16x16*)(vp + (size_t)(0 * 16 + lane15) * Sdim), o0);
        o1 = bwmma(pa, *(const bf16x16*)(vp + (size_t)(1 * 16 + lane15) * Sdim), o1);
        o2 = bwmma(pa, *(const bf16x16*)(vp + (size_t)(2 * 16 + lane15) * Sdim), o2);
        o3 = bwmma(pa, *(const bf16x16*)(vp + (size_t)(3 * 16 + lane15) * Sdim), o3);
    }
    // ---- normalize and store (row-major tokens, bf16)
#pragma unroll
    for (int r = 0; r < 8; ++r) {
        float inv = 1.f / rowsum[r];
        size_t base = (size_t)(b * Sdim + q0 + r + 8 * kh) * Dn + h * DKn + lane15;
        O[base + 0]  = (bf16)(o0[r] * inv);
        O[base + 16] = (bf16)(o1[r] * inv);
        O[base + 32] = (bf16)(o2[r] * inv);
        O[base + 48] = (bf16)(o3[r] * inv);
    }
}

// ---------------------------------------------------------------------------
// Host orchestration
// ---------------------------------------------------------------------------
static void launch_gemm(int epi, const bf16* A, const bf16* Bt, int M, int N, int K,
                        float* oF, bf16* oB, const float* bias, const float* res,
                        int S_, int H_, int DK_, hipStream_t st) {
    dim3 g(M / 64, N / 256), blk(256);
    switch (epi) {
        case EPI_F32:            gemm_bf16_kernel<EPI_F32><<<g, blk, 0, st>>>(A, Bt, M, N, K, oF, oB, bias, res, S_, H_, DK_); break;
        case EPI_BIAS_RES_F32:   gemm_bf16_kernel<EPI_BIAS_RES_F32><<<g, blk, 0, st>>>(A, Bt, M, N, K, oF, oB, bias, res, S_, H_, DK_); break;
        case EPI_BF16:           gemm_bf16_kernel<EPI_BF16><<<g, blk, 0, st>>>(A, Bt, M, N, K, oF, oB, bias, res, S_, H_, DK_); break;
        case EPI_BIAS_GELU_BF16: gemm_bf16_kernel<EPI_BIAS_GELU_BF16><<<g, blk, 0, st>>>(A, Bt, M, N, K, oF, oB, bias, res, S_, H_, DK_); break;
        default:                 gemm_bf16_kernel<EPI_BF16_VT><<<g, blk, 0, st>>>(A, Bt, M, N, K, oF, oB, bias, res, S_, H_, DK_); break;
    }
}

extern "C" void kernel_launch(void* const* d_in, const int* in_sizes, int n_in,
                              void* d_out, int out_size, void* d_ws, size_t ws_size,
                              hipStream_t stream) {
    (void)in_sizes; (void)n_in; (void)out_size; (void)ws_size;
    const int*   ids  = (const int*)d_in[0];
    const float* emb  = (const float*)d_in[1];
    const float* Wq   = (const float*)d_in[2];
    const float* Wk   = (const float*)d_in[3];
    const float* Wv   = (const float*)d_in[4];
    const float* Wo   = (const float*)d_in[5];
    const float* bo   = (const float*)d_in[6];
    const float* ln1w = (const float*)d_in[7];
    const float* ln1b = (const float*)d_in[8];
    const float* ln2w = (const float*)d_in[9];
    const float* ln2b = (const float*)d_in[10];
    const float* W1   = (const float*)d_in[11];
    const float* b1   = (const float*)d_in[12];
    const float* W2   = (const float*)d_in[13];
    const float* b2   = (const float*)d_in[14];
    const float* lnfw = (const float*)d_in[15];
    const float* lnfb = (const float*)d_in[16];
    const float* Wout = (const float*)d_in[17];
    float* logits = (float*)d_out;

    constexpr int Vv = 32000, D = 1024, Hh = 16, DFF = 4096, Ll = 6, Bb = 2, Ss = 2048, DKk = 64;
    constexpr int M = Bb * Ss;  // 4096 tokens

    // ---- workspace carve-up (256B aligned), ~309 MB total
    char*  ws  = (char*)d_ws;
    size_t off = 0;
    auto alloc = [&](size_t bytes) -> void* {
        void* p = ws + off;
        off += (bytes + 255) & ~(size_t)255;
        return p;
    };
    bf16*  wtq  = (bf16*)alloc((size_t)Ll * D * D * 2);
    bf16*  wtk  = (bf16*)alloc((size_t)Ll * D * D * 2);
    bf16*  wtv  = (bf16*)alloc((size_t)Ll * D * D * 2);
    bf16*  wto  = (bf16*)alloc((size_t)Ll * D * D * 2);
    bf16*  wt1  = (bf16*)alloc((size_t)Ll * D * DFF * 2);   // [l][DFF][D]
    bf16*  wt2  = (bf16*)alloc((size_t)Ll * DFF * D * 2);   // [l][D][DFF]
    bf16*  wtout= (bf16*)alloc((size_t)Vv * D * 2);         // [V][D]
    float* xbuf = (float*)alloc((size_t)M * D * 4);
    bf16*  hbuf = (bf16*) alloc((size_t)M * D * 2);
    bf16*  qb   = (bf16*) alloc((size_t)M * D * 2);
    bf16*  kb   = (bf16*) alloc((size_t)M * D * 2);
    bf16*  vtb  = (bf16*) alloc((size_t)M * D * 2);         // [B][H][DK][S]
    bf16*  ob   = (bf16*) alloc((size_t)M * D * 2);
    bf16*  ff1  = (bf16*) alloc((size_t)M * DFF * 2);

    // ---- weights: f32 -> bf16, transposed per matrix
    {
        long t;
        t = (long)Ll * D * D;
        cvt_transpose_kernel<<<(unsigned)(t / 256), 256, 0, stream>>>(Wq, wtq, D, D, t);
        cvt_transpose_kernel<<<(unsigned)(t / 256), 256, 0, stream>>>(Wk, wtk, D, D, t);
        cvt_transpose_kernel<<<(unsigned)(t / 256), 256, 0, stream>>>(Wv, wtv, D, D, t);
        cvt_transpose_kernel<<<(unsigned)(t / 256), 256, 0, stream>>>(Wo, wto, D, D, t);
        t = (long)Ll * D * DFF;
        cvt_transpose_kernel<<<(unsigned)(t / 256), 256, 0, stream>>>(W1, wt1, D, DFF, t);
        cvt_transpose_kernel<<<(unsigned)(t / 256), 256, 0, stream>>>(W2, wt2, DFF, D, t);
        t = (long)Vv * D;
        cvt_transpose_kernel<<<(unsigned)(t / 256), 256, 0, stream>>>(Wout, wtout, D, Vv, t);
    }

    // ---- embedding + positional encoding
    embed_pe_kernel<<<M, 256, 0, stream>>>(ids, emb, xbuf, Ss, D);

    // ---- transformer layers
    for (int l = 0; l < Ll; ++l) {
        const bf16* wq = wtq + (size_t)l * D * D;
        const bf16* wk = wtk + (size_t)l * D * D;
        const bf16* wv = wtv + (size_t)l * D * D;
        const bf16* wo = wto + (size_t)l * D * D;
        const bf16* w1 = wt1 + (size_t)l * D * DFF;
        const bf16* w2 = wt2 + (size_t)l * DFF * D;

        layernorm_bf16_kernel<<<M, 256, 0, stream>>>(xbuf, ln1w + l * D, ln1b + l * D, hbuf, D);

        launch_gemm(EPI_BF16,    hbuf, wq, M, D, D, nullptr, qb,  nullptr, nullptr, Ss, Hh, DKk, stream);
        launch_gemm(EPI_BF16,    hbuf, wk, M, D, D, nullptr, kb,  nullptr, nullptr, Ss, Hh, DKk, stream);
        launch_gemm(EPI_BF16_VT, hbuf, wv, M, D, D, nullptr, vtb, nullptr, nullptr, Ss, Hh, DKk, stream);

        flash_attn_kernel<<<dim3(Ss / 128, Bb * Hh), 256, 0, stream>>>(
            qb, kb, vtb, ob, Ss, Hh, DKk, D, 0.125f);

        launch_gemm(EPI_BIAS_RES_F32, ob, wo, M, D, D, xbuf, nullptr, bo + l * D, xbuf, Ss, Hh, DKk, stream);

        layernorm_bf16_kernel<<<M, 256, 0, stream>>>(xbuf, ln2w + l * D, ln2b + l * D, hbuf, D);

        launch_gemm(EPI_BIAS_GELU_BF16, hbuf, w1, M, DFF, D, nullptr, ff1, b1 + l * DFF, nullptr, Ss, Hh, DKk, stream);
        launch_gemm(EPI_BIAS_RES_F32,   ff1,  w2, M, D, DFF, xbuf, nullptr, b2 + l * D, xbuf, Ss, Hh, DKk, stream);
    }

    // ---- final LN + vocab projection
    layernorm_bf16_kernel<<<M, 256, 0, stream>>>(xbuf, lnfw, lnfb, hbuf, D);
    launch_gemm(EPI_F32, hbuf, wtout, M, Vv, D, logits, nullptr, nullptr, nullptr, Ss, Hh, DKk, stream);
}
